// RGCN_10282151706760
// MI455X (gfx1250) — matrix operand order repaired and checked by reading, not compiled, not run on hardware
//
#include <hip/hip_runtime.h>
#include <hip/hip_bf16.h>
#include <stdint.h>

// ---------------- problem constants (match reference) ----------------
#define N_NODES 50000
#define N_EDGES 800000
#define NREL    8
#define NGRAPH  128
#define F_IN    384   // 6 * 64
#define F_HID   256   // LAYER == OUT == 256

typedef __attribute__((ext_vector_type(16))) __bf16   v16bf;
typedef __attribute__((ext_vector_type(8)))  float    v8f;
typedef __attribute__((ext_vector_type(4)))  uint32_t v4u;
typedef __attribute__((ext_vector_type(2)))  uint32_t v2u;
typedef __attribute__((ext_vector_type(4)))  float    v4f;

union FragAB { v16bf bf; v4u q[2]; };

__device__ __forceinline__ uint16_t f2bf(float f) {
  uint32_t u = __float_as_uint(f);
  uint32_t r = u + 0x7fffu + ((u >> 16) & 1u);   // round-to-nearest-even
  return (uint16_t)(r >> 16);
}
__device__ __forceinline__ float bf2f(uint32_t h16) {
  return __uint_as_float(h16 << 16);
}
__device__ __forceinline__ float sigmoidf(float x) {
  return 1.0f / (1.0f + expf(-x));
}

// ---------------- generic zero ----------------
__global__ void k_zero(float* __restrict__ p, long n) {
  long t = (long)blockIdx.x * blockDim.x + threadIdx.x;
  if (t < n) p[t] = 0.0f;
}

// ---------------- embedding gather: h0[n, tbl*64+j] = emb_tbl[x[n,tbl]][j] (bf16) ----
__global__ void k_embed(const int* __restrict__ x,
                        const float* __restrict__ e0, const float* __restrict__ e1,
                        const float* __restrict__ e2, const float* __restrict__ e3,
                        const float* __restrict__ e4, const float* __restrict__ e5,
                        uint16_t* __restrict__ h0) {
  long t = (long)blockIdx.x * blockDim.x + threadIdx.x;
  if (t >= (long)N_NODES * F_IN) return;
  int n = (int)(t / F_IN), f = (int)(t % F_IN);
  int tbl = f >> 6, j = f & 63;
  const float* tab;
  switch (tbl) {
    case 0: tab = e0; break; case 1: tab = e1; break; case 2: tab = e2; break;
    case 3: tab = e3; break; case 4: tab = e4; break; default: tab = e5; break;
  }
  int idx = x[(size_t)n * 6 + tbl];
  h0[t] = f2bf(tab[(size_t)idx * 64 + j]);
}

// ---------------- transpose + f32->bf16: dst[Nc,K] = src[K,Nc] ----------------
__global__ void k_transpose_bf16(const float* __restrict__ src, uint16_t* __restrict__ dst,
                                 int K, int Nc, long srcSlab, long dstSlab) {
  long z = blockIdx.z;
  src += z * srcSlab; dst += z * dstSlab;
  long t = (long)blockIdx.x * blockDim.x + threadIdx.x;
  if (t >= (long)K * Nc) return;
  int k = (int)(t / Nc), n = (int)(t % Nc);
  dst[(size_t)n * K + k] = f2bf(src[t]);
}

// ---------------- WMMA bf16 GEMM: C[M,Nc] = A[M,K] * B[K,Nc]  (B given as BT[Nc,K]) ----
// Block = 256 threads = 8 waves. Block tile 128x64. Wave tile 16x64 (4 wmma accum).
// B strip (64 x K) resident in LDS (async-loaded once); A double-buffered with
// GLOBAL_LOAD_ASYNC_TO_LDS_B128 (ASYNCcnt) overlapping the WMMA pipeline.
template<bool BF16OUT, int K>
__global__ __launch_bounds__(256)
void gemm_bf16(const uint16_t* __restrict__ A, const uint16_t* __restrict__ BT,
               void* __restrict__ Cv, int M, int Nc) {
  constexpr int LDB = K + 8;              // padded halves per B row (K mult of 32 -> 16B aligned)
  constexpr int NIT = K / 32;
  __shared__ uint16_t ldsA[2][128 * 40];  // 128 rows x 32 halves, stride 40 (pad)
  __shared__ uint16_t ldsB[64 * LDB];     // resident B strip

  const int tid  = threadIdx.x;
  const int wave = tid >> 5;
  const int lane = tid & 31;
  const int lhi  = lane >> 4;             // 0/1 (half-wave)
  const int llo  = lane & 15;
  const int tileM = blockIdx.x * 128;
  const int tileN = blockIdx.y * 64;

  v8f acc[4] = {};

  // ---- staging assignments (constant over k loop) ----
  const int ar = tid >> 1, ah = tid & 1;  // A: row 0..127, 32B half-row chunk
  int agr = tileM + ar; if (agr >= M) agr = M - 1;   // clamp (keeps EXEC full for wmma)
  const uint16_t* aSrc = A + (size_t)agr * K + ah * 16;
  uint32_t aDst[2];
  aDst[0] = (uint32_t)(uintptr_t)&ldsA[0][ar * 40 + ah * 16];
  aDst[1] = (uint32_t)(uintptr_t)&ldsA[1][ar * 40 + ah * 16];

  // ---- async-load resident B strip: 64 rows x K halves, 16B chunks ----
  constexpr int ROWCH = K / 8;                       // 16B chunks per row
#pragma unroll
  for (int c = 0; c < (64 * ROWCH) / 256; ++c) {
    int id  = tid + c * 256;
    int row = id / ROWCH, cc = id % ROWCH;
    uint32_t ld = (uint32_t)(uintptr_t)&ldsB[row * LDB + cc * 8];
    uint64_t ga = (uint64_t)(uintptr_t)(BT + (size_t)(tileN + row) * K + cc * 8);
    asm volatile("global_load_async_to_lds_b128 %0, %1, off"
                 :: "v"(ld), "v"(ga) : "memory");
  }
  // ---- prologue: async-load first A tile ----
  {
    uint64_t ga = (uint64_t)(uintptr_t)aSrc;
    asm volatile("global_load_async_to_lds_b128 %0, %1, off"
                 :: "v"(aDst[0]), "v"(ga) : "memory");
    asm volatile("global_load_async_to_lds_b128 %0, %1, off offset:16"
                 :: "v"(aDst[0]), "v"(ga) : "memory");
  }

#pragma unroll
  for (int i = 0; i < NIT; ++i) {
    if (i + 1 < NIT) {
      // issue next A tile into the other buffer (readers of it synced last iter)
      uint64_t ga = (uint64_t)(uintptr_t)(aSrc + (i + 1) * 32);
      asm volatile("global_load_async_to_lds_b128 %0, %1, off"
                   :: "v"(aDst[(i + 1) & 1]), "v"(ga) : "memory");
      asm volatile("global_load_async_to_lds_b128 %0, %1, off offset:16"
                   :: "v"(aDst[(i + 1) & 1]), "v"(ga) : "memory");
      if (i + 2 < NIT)
        __builtin_prefetch(aSrc + (i + 2) * 32, 0, 1);  // warm L2 one stage further ahead
      // async ops complete in order: <=2 outstanding -> current tile (and B strip) landed
      asm volatile("s_wait_asynccnt 0x2" ::: "memory");
    } else {
      asm volatile("s_wait_asynccnt 0x0" ::: "memory");
    }
    __syncthreads();

    // A fragment: lane = row M=llo of wave strip; K = lhi*8+e (e<8), 16+lhi*8+e (e>=8)
    FragAB a;
    const uint16_t* ap = &ldsA[i & 1][(wave * 16 + llo) * 40 + lhi * 8];
    a.q[0] = *(const v4u*)ap;
    a.q[1] = *(const v4u*)(ap + 16);

    // B fragments: lane = col N=llo; K = k0 + lhi*8+e (e<8), k0 + 16+lhi*8+e (e>=8)
    FragAB bf[4];
#pragma unroll
    for (int j = 0; j < 4; ++j) {
      const uint16_t* bp = &ldsB[(j * 16 + llo) * LDB + i * 32 + lhi * 8];
      bf[j].q[0] = *(const v4u*)bp;
      bf[j].q[1] = *(const v4u*)(bp + 16);
    }
#pragma unroll
    for (int j = 0; j < 4; ++j)
      acc[j] = __builtin_amdgcn_wmma_f32_16x16x32_bf16(
          false, a.bf, false, bf[j].bf, (short)0, acc[j], false, false);

    __syncthreads();   // everyone done reading ldsA[i&1] before it is re-filled
  }

  // C/D layout: VGPR v, lane l -> M = v + 8*(l>=16), N = l&15
  float*    Cf = (float*)Cv;
  uint16_t* Cb = (uint16_t*)Cv;
#pragma unroll
  for (int j = 0; j < 4; ++j) {
    int col = tileN + j * 16 + llo;
#pragma unroll
    for (int v = 0; v < 8; ++v) {
      int row = tileM + wave * 16 + v + 8 * lhi;
      if (row < M) {
        if (BF16OUT) Cb[(size_t)row * Nc + col] = f2bf(acc[j][v]);
        else         Cf[(size_t)row * Nc + col] = acc[j][v];
      }
    }
  }
}

// ---------------- edge count per (dst, rel) ----------------
__global__ void k_count(const int* __restrict__ dst, const int* __restrict__ et,
                        float* __restrict__ cnt) {
  long e = (long)blockIdx.x * blockDim.x + threadIdx.x;
  if (e >= N_EDGES) return;
  atomicAdd(&cnt[(size_t)dst[e] * NREL + et[e]], 1.0f);
}

// ---------------- scatter one relation's messages into acc (mean aggr) ----------
__global__ void k_scatter(const uint16_t* __restrict__ msg, const int* __restrict__ src,
                          const int* __restrict__ dst, const int* __restrict__ et,
                          const float* __restrict__ cnt, float* __restrict__ acc, int rel) {
  long t = (long)blockIdx.x * blockDim.x + threadIdx.x;
  long e = t >> 6;
  if (e >= N_EDGES) return;
  if (et[e] != rel) return;
  int c0 = (int)(t & 63) * 4;
  int s = src[e], d = dst[e];
  float norm = 1.0f / fmaxf(cnt[(size_t)d * NREL + rel], 1.0f);
  v2u raw = *(const v2u*)(msg + (size_t)s * F_HID + c0);
  float* ap = acc + (size_t)d * F_HID + c0;
  atomicAdd(ap + 0, bf2f(raw.x & 0xffffu) * norm);
  atomicAdd(ap + 1, bf2f(raw.x >> 16)     * norm);
  atomicAdd(ap + 2, bf2f(raw.y & 0xffffu) * norm);
  atomicAdd(ap + 3, bf2f(raw.y >> 16)     * norm);
}

// ---------------- h = sigmoid(acc + root + b); write bf16 (+ optional f32) --------
__global__ void k_finalize(const float* __restrict__ acc, const uint16_t* __restrict__ rootc,
                           const float* __restrict__ bias, uint16_t* __restrict__ out_bf,
                           float* __restrict__ out_f32, int storef) {
  long t = (long)blockIdx.x * blockDim.x + threadIdx.x;
  if (t >= (long)N_NODES * F_HID) return;
  int c = (int)(t & (F_HID - 1));
  float s = sigmoidf(acc[t] + bf2f(rootc[t]) + bias[c]);
  out_bf[t] = f2bf(s);
  if (storef) out_f32[t] = s;
}

// ---------------- BatchNorm stats: per-channel sum / sumsq over all N rows -------
__global__ void k_bnstats(const float* __restrict__ hg, float* __restrict__ sums) {
  int c = threadIdx.x;                       // 256 channels
  long n0 = (long)blockIdx.x * 250;
  float s = 0.f, s2 = 0.f;
  for (int i = 0; i < 250; ++i) {
    long n = n0 + i;
    if (n < N_NODES) {
      float v = hg[n * F_HID + c];
      s += v; s2 += v * v;
    }
  }
  atomicAdd(&sums[c], s);
  atomicAdd(&sums[F_HID + c], s2);
}

// ---------------- gate: BN + ReLU + dot(Wg2) + segment-max (ordered-uint) --------
__global__ __launch_bounds__(256)
void k_gate(const float* __restrict__ hg, const float* __restrict__ sums,
            const float* __restrict__ gamma, const float* __restrict__ beta,
            const float* __restrict__ Wg2, const float* __restrict__ bg2,
            const int* __restrict__ batch, float* __restrict__ gate,
            unsigned* __restrict__ gmax) {
  int wave = threadIdx.x >> 5, lane = threadIdx.x & 31;
  long n = (long)blockIdx.x * 8 + wave;
  if (n >= N_NODES) return;
  const float invN = 1.0f / (float)N_NODES;
  float part = 0.f;
#pragma unroll
  for (int k = 0; k < 8; ++k) {
    int c = lane + 32 * k;
    float mu  = sums[c] * invN;
    float var = sums[F_HID + c] * invN - mu * mu;
    float v = (hg[n * F_HID + c] - mu) * rsqrtf(var + 1e-5f) * gamma[c] + beta[c];
    v = fmaxf(v, 0.f);
    part += v * Wg2[c];
  }
  for (int off = 16; off; off >>= 1) part += __shfl_down(part, off);
  if (lane == 0) {
    float g = part + bg2[0];
    gate[n] = g;
    unsigned u = __float_as_uint(g);
    u ^= (u & 0x80000000u) ? 0xFFFFFFFFu : 0x80000000u;   // order-preserving key
    atomicMax(&gmax[batch[n]], u);
  }
}

// ---------------- e = exp(gate - gmax[g]); den[g] += e ----------------
__global__ void k_expsum(const float* __restrict__ gate, const unsigned* __restrict__ gmax,
                         const int* __restrict__ batch, float* __restrict__ ew,
                         float* __restrict__ den) {
  long n = (long)blockIdx.x * blockDim.x + threadIdx.x;
  if (n >= N_NODES) return;
  int g = batch[n];
  unsigned k = gmax[g];
  unsigned bits = (k & 0x80000000u) ? (k ^ 0x80000000u) : ~k;
  float m = __uint_as_float(bits);
  float e = expf(gate[n] - m);
  ew[n] = e;
  atomicAdd(&den[g], e);
}

// ---------------- pooled[g,:] += alpha * h2[n,:] ----------------
__global__ void k_pool(const float* __restrict__ h2f, const float* __restrict__ ew,
                       const float* __restrict__ den, const int* __restrict__ batch,
                       float* __restrict__ pooled) {
  long t = (long)blockIdx.x * blockDim.x + threadIdx.x;
  long n = t >> 6;
  if (n >= N_NODES) return;
  int c0 = (int)(t & 63) * 4;
  int g = batch[n];
  float alpha = ew[n] / den[g];
  v4f v = *(const v4f*)(h2f + (size_t)n * F_HID + c0);
  float* pp = pooled + (size_t)g * F_HID + c0;
  atomicAdd(pp + 0, alpha * v.x);
  atomicAdd(pp + 1, alpha * v.y);
  atomicAdd(pp + 2, alpha * v.z);
  atomicAdd(pp + 3, alpha * v.w);
}

// ---------------- head: out[g] = sigmoid(pooled[g,:] . Wgl + bgl) ----------------
__global__ __launch_bounds__(256)
void k_head(const float* __restrict__ pooled, const float* __restrict__ Wgl,
            const float* __restrict__ bgl, float* __restrict__ out) {
  int wave = threadIdx.x >> 5, lane = threadIdx.x & 31;
  int g = blockIdx.x * 8 + wave;
  if (g >= NGRAPH) return;
  float part = 0.f;
#pragma unroll
  for (int k = 0; k < 8; ++k) {
    int c = lane + 32 * k;
    part += pooled[(size_t)g * F_HID + c] * Wgl[c];
  }
  for (int off = 16; off; off >>= 1) part += __shfl_down(part, off);
  if (lane == 0) out[g] = sigmoidf(part + bgl[0]);
}

// ======================================================================
extern "C" void kernel_launch(void* const* d_in, const int* in_sizes, int n_in,
                              void* d_out, int out_size, void* d_ws, size_t ws_size,
                              hipStream_t stream) {
  // ---- inputs (setup_inputs order) ----
  const int*   x      = (const int*)d_in[0];
  const int*   eidx   = (const int*)d_in[1];      // [2,E]
  const int*   etype  = (const int*)d_in[2];
  const int*   batch  = (const int*)d_in[3];
  const float* emb[6] = {(const float*)d_in[4], (const float*)d_in[5], (const float*)d_in[6],
                         (const float*)d_in[7], (const float*)d_in[8], (const float*)d_in[9]};
  const float* W1    = (const float*)d_in[10];
  const float* root1 = (const float*)d_in[11];
  const float* b1    = (const float*)d_in[12];
  const float* W2    = (const float*)d_in[13];
  const float* root2 = (const float*)d_in[14];
  const float* b2    = (const float*)d_in[15];
  const float* Wg1   = (const float*)d_in[16];
  // d_in[17] = bg1: constant per-channel shift -> cancels in BatchNorm, unused
  const float* gamma = (const float*)d_in[18];
  const float* beta  = (const float*)d_in[19];
  const float* Wg2   = (const float*)d_in[20];
  const float* bg2   = (const float*)d_in[21];
  const float* Wgl   = (const float*)d_in[22];
  const float* bgl   = (const float*)d_in[23];
  float* out = (float*)d_out;

  const int* esrc = eidx;
  const int* edst = eidx + N_EDGES;

  // ---- workspace carve ----
  char* base = (char*)d_ws;
  size_t off = 0;
  auto alloc = [&](size_t bytes) -> void* {
    void* p = base + off;
    off = (off + bytes + 255) & ~(size_t)255;
    return p;
  };
  uint16_t* h0    = (uint16_t*)alloc((size_t)N_NODES * F_IN * 2);          // bf16 input feats
  uint16_t* W1T   = (uint16_t*)alloc((size_t)(NREL + 1) * F_HID * F_IN * 2);  // slab 8 = root1
  uint16_t* W2T   = (uint16_t*)alloc((size_t)(NREL + 1) * F_HID * F_HID * 2); // slab 8 = root2
  uint16_t* Wg1T  = (uint16_t*)alloc((size_t)F_HID * F_HID * 2);
  uint16_t* msg   = (uint16_t*)alloc((size_t)N_NODES * F_HID * 2);         // one relation's messages
  uint16_t* rootc = (uint16_t*)alloc((size_t)N_NODES * F_HID * 2);         // root contribution
  float*    acc   = (float*)alloc((size_t)N_NODES * F_HID * 4);            // conv accum / gate hg
  float*    cnt   = (float*)alloc((size_t)N_NODES * NREL * 4);
  uint16_t* h1b   = (uint16_t*)alloc((size_t)N_NODES * F_HID * 2);
  uint16_t* h2b   = (uint16_t*)alloc((size_t)N_NODES * F_HID * 2);
  float*    h2f   = (float*)alloc((size_t)N_NODES * F_HID * 4);
  float*    gate  = (float*)alloc((size_t)N_NODES * 4);
  float*    ew    = (float*)alloc((size_t)N_NODES * 4);
  float*    sums  = (float*)alloc(2 * F_HID * 4);
  unsigned* gmax  = (unsigned*)alloc(NGRAPH * 4);
  float*    den   = (float*)alloc(NGRAPH * 4);
  float*    pooled= (float*)alloc((size_t)NGRAPH * F_HID * 4);

  auto blocks = [](long n) { return dim3((unsigned)((n + 255) / 256)); };
  const long W1slab = (long)F_HID * F_IN;    // bf16 elems per transposed slab
  const long W2slab = (long)F_HID * F_HID;

  // ---- edge degree counts (shared by both convs) ----
  k_zero<<<blocks((long)N_NODES * NREL), 256, 0, stream>>>(cnt, (long)N_NODES * NREL);
  k_count<<<blocks(N_EDGES), 256, 0, stream>>>(edst, etype, cnt);

  // ---- embedding gather ----
  k_embed<<<blocks((long)N_NODES * F_IN), 256, 0, stream>>>(
      x, emb[0], emb[1], emb[2], emb[3], emb[4], emb[5], h0);

  // ---- weight transpose/convert (root folded in as slab NREL) ----
  {
    dim3 g1(blocks((long)F_IN * F_HID).x, 1, NREL);
    k_transpose_bf16<<<g1, 256, 0, stream>>>(W1, W1T, F_IN, F_HID,
                                             (long)F_IN * F_HID, W1slab);
    k_transpose_bf16<<<dim3(blocks((long)F_IN * F_HID).x, 1, 1), 256, 0, stream>>>(
        root1, W1T + NREL * W1slab, F_IN, F_HID, 0, 0);
    dim3 g2(blocks((long)F_HID * F_HID).x, 1, NREL);
    k_transpose_bf16<<<g2, 256, 0, stream>>>(W2, W2T, F_HID, F_HID,
                                             (long)F_HID * F_HID, W2slab);
    k_transpose_bf16<<<dim3(blocks((long)F_HID * F_HID).x, 1, 1), 256, 0, stream>>>(
        root2, W2T + NREL * W2slab, F_HID, F_HID, 0, 0);
    k_transpose_bf16<<<dim3(blocks((long)F_HID * F_HID).x, 1, 1), 256, 0, stream>>>(
        Wg1, Wg1T, F_HID, F_HID, 0, 0);
  }

  dim3 gemmGrid((N_NODES + 127) / 128, F_HID / 64, 1);
  dim3 scatGrid((unsigned)(((long)N_EDGES * 64 + 255) / 256));

  // ---- conv1: root GEMM + per-relation GEMM->scatter, then sigmoid ----
  gemm_bf16<true, F_IN><<<gemmGrid, 256, 0, stream>>>(h0, W1T + NREL * W1slab, (void*)rootc,
                                                      N_NODES, F_HID);
  k_zero<<<blocks((long)N_NODES * F_HID), 256, 0, stream>>>(acc, (long)N_NODES * F_HID);
  for (int r = 0; r < NREL; ++r) {
    gemm_bf16<true, F_IN><<<gemmGrid, 256, 0, stream>>>(h0, W1T + (long)r * W1slab, (void*)msg,
                                                        N_NODES, F_HID);
    k_scatter<<<scatGrid, 256, 0, stream>>>(msg, esrc, edst, etype, cnt, acc, r);
  }
  k_finalize<<<blocks((long)N_NODES * F_HID), 256, 0, stream>>>(acc, rootc, b1, h1b, nullptr, 0);

  // ---- conv2 ----
  gemm_bf16<true, F_HID><<<gemmGrid, 256, 0, stream>>>(h1b, W2T + NREL * W2slab, (void*)rootc,
                                                       N_NODES, F_HID);
  k_zero<<<blocks((long)N_NODES * F_HID), 256, 0, stream>>>(acc, (long)N_NODES * F_HID);
  for (int r = 0; r < NREL; ++r) {
    gemm_bf16<true, F_HID><<<gemmGrid, 256, 0, stream>>>(h1b, W2T + (long)r * W2slab, (void*)msg,
                                                         N_NODES, F_HID);
    k_scatter<<<scatGrid, 256, 0, stream>>>(msg, esrc, edst, etype, cnt, acc, r);
  }
  k_finalize<<<blocks((long)N_NODES * F_HID), 256, 0, stream>>>(acc, rootc, b2, h2b, h2f, 1);

  // ---- global attention pooling ----
  gemm_bf16<false, F_HID><<<gemmGrid, 256, 0, stream>>>(h2b, Wg1T, (void*)acc,  // acc = hg (f32)
                                                        N_NODES, F_HID);
  k_zero<<<blocks(2 * F_HID), 256, 0, stream>>>(sums, 2 * F_HID);
  k_bnstats<<<dim3((N_NODES + 249) / 250), 256, 0, stream>>>(acc, sums);
  k_zero<<<blocks(NGRAPH), 256, 0, stream>>>((float*)gmax, NGRAPH);       // key 0 == -inf
  k_gate<<<dim3((N_NODES + 7) / 8), 256, 0, stream>>>(acc, sums, gamma, beta, Wg2, bg2,
                                                      batch, gate, gmax);
  k_zero<<<blocks(NGRAPH), 256, 0, stream>>>(den, NGRAPH);
  k_expsum<<<blocks(N_NODES), 256, 0, stream>>>(gate, gmax, batch, ew, den);
  k_zero<<<blocks((long)NGRAPH * F_HID), 256, 0, stream>>>(pooled, (long)NGRAPH * F_HID);
  k_pool<<<dim3((unsigned)(((long)N_NODES * 64 + 255) / 256)), 256, 0, stream>>>(
      h2f, ew, den, batch, pooled);

  // ---- head ----
  k_head<<<dim3((NGRAPH + 7) / 8), 256, 0, stream>>>(pooled, Wgl, bgl, out);
}
